// TimeSeries_16131897164029
// MI455X (gfx1250) — matrix operand order repaired and checked by reading, not compiled, not run on hardware
//
#include <hip/hip_runtime.h>
#include <hip/hip_bf16.h>
#include <math.h>

typedef __attribute__((ext_vector_type(16))) __bf16 v16bf;
typedef __attribute__((ext_vector_type(8)))  __bf16 v8bf;
typedef __attribute__((ext_vector_type(8)))  float  v8f;
typedef __attribute__((ext_vector_type(4)))  unsigned int v4u;

#define B_   128
#define T_   256
#define I_   128
#define H_   256
#define D_   8
#define G3   768          // 3*H
#define EPS_ 1e-4f
#define NSPLIT 4          // workgroups per direction in the recurrence
#define LDW  264          // staged W_hh K-row stride in bf16 (256 + 8 pad -> conflict-free b128)

// ---------------- workspace layout (bytes, all 256B aligned) ----------------
#define WS_CTR   0                                      // 2 ints
#define WS_XBF   256                                    // bf16 x, [T][B][I]
#define WS_WIH   (WS_XBF + 2L*T_*B_*I_)                 // bf16 w_ih f then b, [768][128]
#define WS_WHH   (WS_WIH + 2L*2*G3*I_)                  // bf16 w_hh f then b, [768][256]
#define WS_HBUF  (WS_WHH + 2L*2*G3*H_)                  // bf16 h ping-pong [2dir][2][128][256]
#define WS_ENC   (WS_HBUF + 2L*2*2*B_*H_)               // bf16 enc [B][T][512]
#define WS_BD    (WS_ENC + 2L*B_*T_*(2*H_))             // f32 bd [B][T][16]
#define WS_MAIN  (WS_BD + 4L*B_*T_*16)                  // f32 [B*D][T]
#define WS_AB    (WS_MAIN + 4L*B_*D_*T_)
#define WS_L     (WS_AB  + 4L*B_*D_*T_)
#define WS_SUB   (WS_L   + 4L*B_*D_*T_)
#define WS_GX    (WS_SUB + 4L*B_*D_*T_)                 // bf16 gx [2][T][768][128]

// ---------------- WMMA fragment loaders (contiguous k order, same for A and B) ----------------
// A 16x32 bf16: lanes 0-15 row=lane, K 0..15; lanes 16-31 row=lane-16, K 16..31
__device__ __forceinline__ v16bf load_a_frag(const __bf16* base, int ld, int row0, int k0) {
  int lane = threadIdx.x & 31;
  int m  = row0 + (lane & 15);
  int kb = k0 + ((lane >> 4) << 4);
  const __bf16* p = base + (size_t)m * ld + kb;
  v8bf lo = *reinterpret_cast<const v8bf*>(p);
  v8bf hi = *reinterpret_cast<const v8bf*>(p + 8);
  v16bf a;
#pragma unroll
  for (int i = 0; i < 8; ++i) { a[i] = lo[i]; a[8 + i] = hi[i]; }
  return a;
}
// B 32x16 bf16 from row-major W[N][K]: lane col n=lane&15; lanes0-15 K=k0..+15, lanes16-31 K=k0+16..+31
__device__ __forceinline__ v16bf load_b_frag(const __bf16* w, int ldk, int k0, int n0) {
  int lane = threadIdx.x & 31;
  int n  = n0 + (lane & 15);
  int kb = k0 + ((lane >> 4) << 4);
  return *reinterpret_cast<const v16bf*>(w + (size_t)n * ldk + kb);
}
// All three gate B-fragments for one k-step from LDS-staged W_hh.
// Explicit ds_load_b128 (asm volatile): cannot be hoisted by LICM (no spill storm),
// guaranteed DS path, one s_wait_dscnt for all six loads.
// Gate g lives 16*g rows further: byte delta = 16*LDW*2 = 8448, 32*LDW*2 = 16896.
__device__ __forceinline__ void lds_b3(const __bf16* sw, int jtl, int k0,
                                       v16bf& wr, v16bf& wz, v16bf& wn) {
  const int lane = threadIdx.x & 31;
  const int nc = lane & 15;
  const int kb = k0 + ((lane >> 4) << 4);
  unsigned a0 = (unsigned)(size_t)(sw + ((size_t)(jtl * 3) * 16 + nc) * LDW + kb);
  v4u r0, r1, z0, z1, n0, n1;
  asm volatile(
      "ds_load_b128 %0, %6\n\t"
      "ds_load_b128 %1, %6 offset:16\n\t"
      "ds_load_b128 %2, %6 offset:8448\n\t"
      "ds_load_b128 %3, %6 offset:8464\n\t"
      "ds_load_b128 %4, %6 offset:16896\n\t"
      "ds_load_b128 %5, %6 offset:16912\n\t"
      "s_wait_dscnt 0x0"
      : "=v"(r0), "=v"(r1), "=v"(z0), "=v"(z1), "=v"(n0), "=v"(n1)
      : "v"(a0));
  union { v4u u[2]; v16bf v; } c0, c1, c2;
  c0.u[0] = r0; c0.u[1] = r1; wr = c0.v;
  c1.u[0] = z0; c1.u[1] = z1; wz = c1.v;
  c2.u[0] = n0; c2.u[1] = n1; wn = c2.v;
}

// ---------------- prep: bf16 conversion + zero init --------------------------
__global__ void prep_kernel(const float* __restrict__ y,
                            const float* __restrict__ wihf, const float* __restrict__ whhf,
                            const float* __restrict__ wihb, const float* __restrict__ whhb,
                            __bf16* __restrict__ xbf, __bf16* __restrict__ wih,
                            __bf16* __restrict__ whh, __bf16* __restrict__ hbuf,
                            int* __restrict__ ctr)
{
  const long nx = (long)T_ * B_ * I_;
  const long nwih = (long)G3 * I_;
  const long nwhh = (long)G3 * H_;
  const long nh = 2L * 2 * B_ * H_;
  const long total = nx + 2 * nwih + 2 * nwhh + nh + 2;
  for (long idx = (long)blockIdx.x * blockDim.x + threadIdx.x; idx < total;
       idx += (long)gridDim.x * blockDim.x) {
    if (idx < nx) {
      int i = (int)(idx % I_);
      long p = idx / I_;
      int b = (int)(p % B_);
      int t = (int)(p / B_);
      xbf[idx] = (__bf16)y[((size_t)b * T_ + t) * I_ + i];   // [T][B][I]
    } else if (idx < nx + 2 * nwih) {
      long k = idx - nx;
      wih[k] = (__bf16)((k < nwih) ? wihf[k] : wihb[k - nwih]);
    } else if (idx < nx + 2 * nwih + 2 * nwhh) {
      long k = idx - nx - 2 * nwih;
      whh[k] = (__bf16)((k < nwhh) ? whhf[k] : whhb[k - nwhh]);
    } else if (idx < nx + 2 * nwih + 2 * nwhh + nh) {
      hbuf[idx - nx - 2 * nwih - 2 * nwhh] = (__bf16)0.f;
    } else {
      ctr[idx - (nx + 2 * nwih + 2 * nwhh + nh)] = 0;
    }
  }
}

// ---------------- gx = x @ w_ih^T for all (dir,t): fully parallel WMMA -------
// gx layout [dir][t][gatecol 768][b 128] bf16 so the recurrence reads 16B/lane.
// One 16x16 tile per wave: A = w_ih rows (gate cols), B(k,n) = x_t[n][k].
__global__ __launch_bounds__(256) void gx_kernel(const __bf16* __restrict__ xbf,
                                                 const __bf16* __restrict__ wih,
                                                 __bf16* __restrict__ gx)
{
  __shared__ float tr[8][16][17];
  const int wave = threadIdx.x >> 5;
  const int lane = threadIdx.x & 31;
  long wtile = (long)blockIdx.x * 8 + wave;     // 2*256*48*8 = 196608 tiles, exact grid
  int btile = (int)(wtile & 7);
  long r1 = wtile >> 3;
  int gtile = (int)(r1 % 48);
  long r2 = r1 / 48;
  int t = (int)(r2 & 255);
  int dir = (int)(r2 >> 8);

  const __bf16* wihd = wih + (size_t)dir * G3 * I_;
  const __bf16* xt = xbf + (size_t)t * B_ * I_;
  v8f acc = {};
#pragma unroll
  for (int ks = 0; ks < 4; ++ks) {
    int k0 = ks * 32;
    v16bf a = load_a_frag(wihd, I_, gtile * 16, k0);
    v16bf b = load_b_frag(xt, I_, k0, btile * 16);
    acc = __builtin_amdgcn_wmma_f32_16x16x32_bf16(false, a, false, b, (short)0, acc, false, false);
  }
#pragma unroll
  for (int g = 0; g < 8; ++g)
    tr[wave][g + ((lane >> 4) << 3)][lane & 15] = acc[g];
  __syncthreads();
  if (lane < 16) {
    v16bf o;
#pragma unroll
    for (int c = 0; c < 16; ++c) o[c] = (__bf16)tr[wave][lane][c];
    int gcol = gtile * 16 + lane;
    __bf16* dst = gx + (((size_t)dir * T_ + t) * G3 + gcol) * B_ + btile * 16;
    *reinterpret_cast<v16bf*>(dst) = o;
  }
}

// ---------------- bidirectional GRU recurrence (WMMA, multi-WGP) -------------
// grid = 8 blocks: blockIdx>>2 = direction, blockIdx&3 = column split.
// W_hh staged in LDS once; per step each wave does 24 v_wmma (h-part only).
__global__ __launch_bounds__(1024) void gru_recurrence(
    const __bf16* __restrict__ gx, const __bf16* __restrict__ whh,
    const float* __restrict__ bih_f, const float* __restrict__ bhh_f,
    const float* __restrict__ bih_b, const float* __restrict__ bhh_b,
    __bf16* __restrict__ hbuf, __bf16* __restrict__ enc, int* __restrict__ ctr)
{
  extern __shared__ char smem[];
  __bf16* sw = (__bf16*)smem;                               // [192][LDW] staged W_hh
  float (*sh_t)[16][17] = (float(*)[16][17])(smem + 192L * LDW * 2);

  const int dir   = blockIdx.x >> 2;
  const int split = blockIdx.x & 3;
  const int wave  = threadIdx.x >> 5;
  const int lane  = threadIdx.x & 31;
  const int mt  = wave & 7;                    // row tile 0..7
  const int jtl = wave >> 3;                   // local hidden tile 0..3
  const int jt  = split * 4 + jtl;             // global hidden tile 0..15

  const __bf16* whhd = whh + (size_t)dir * G3 * H_;
  const float* bih = dir ? bih_b : bih_f;
  const float* bhh = dir ? bhh_b : bhh_f;
  __bf16* hb = hbuf + (size_t)dir * 2 * B_ * H_;
  const __bf16* gxd = gx + (size_t)dir * T_ * G3 * B_;
  int* cdir = ctr + dir;

  // stage this WG's 192 gate-columns of W_hh into LDS (row l = (jl*3+g)*16+nc)
  for (int e = threadIdx.x; e < 192 * 32; e += 1024) {
    int l  = e >> 5;
    int ch = e & 31;                           // 8-bf16 chunk of the K row
    int nc = l & 15;
    int gg = (l >> 4) % 3;
    int jl = (l >> 4) / 3;
    int gcol = gg * H_ + (split * 4 + jl) * 16 + nc;
    *reinterpret_cast<v8bf*>(sw + (size_t)l * LDW + ch * 8) =
        *reinterpret_cast<const v8bf*>(whhd + (size_t)gcol * H_ + ch * 8);
  }
  __syncthreads();

  const int col = jt * 16 + (lane & 15);       // hidden column of this lane (C layout: N=lane&15)
  const float br2 = bih[col] + bhh[col];
  const float bz2 = bih[H_ + col] + bhh[H_ + col];
  const float bnx = bih[2 * H_ + col];
  const float bnh = bhh[2 * H_ + col];
  const int rbase = mt * 16 + ((lane >> 4) << 3);  // first C row held by this lane

  for (int t = 0; t < T_; ++t) {
    const int ta = dir ? (T_ - 1 - t) : t;
    const __bf16* hc = hb + (size_t)(t & 1) * B_ * H_;
    __bf16* hn = hb + (size_t)((t + 1) & 1) * B_ * H_;

    v8f ar = {}, az = {}, anh = {};
#pragma unroll
    for (int ks = 0; ks < 8; ++ks) {                      // h @ w_hh^T, K=256
      int k0 = ks * 32;
      v16bf a = load_a_frag(hc, H_, mt * 16, k0);
      v16bf wr, wz, wn;
      lds_b3(sw, jtl, k0, wr, wz, wn);
      ar  = __builtin_amdgcn_wmma_f32_16x16x32_bf16(false, a, false, wr, (short)0, ar,  false, false);
      az  = __builtin_amdgcn_wmma_f32_16x16x32_bf16(false, a, false, wz, (short)0, az,  false, false);
      anh = __builtin_amdgcn_wmma_f32_16x16x32_bf16(false, a, false, wn, (short)0, anh, false, false);
    }
    // precomputed input-gate contributions: 16B per gate per lane
    const __bf16* gxt = gxd + (size_t)ta * G3 * B_;
    v8bf gr = *reinterpret_cast<const v8bf*>(gxt + (size_t)(0 * H_ + col) * B_ + rbase);
    v8bf gz = *reinterpret_cast<const v8bf*>(gxt + (size_t)(1 * H_ + col) * B_ + rbase);
    v8bf gn = *reinterpret_cast<const v8bf*>(gxt + (size_t)(2 * H_ + col) * B_ + rbase);
#pragma unroll
    for (int g = 0; g < 8; ++g) {
      int row = rbase + g;
      float r = 1.f / (1.f + __expf(-(ar[g] + (float)gr[g] + br2)));
      float z = 1.f / (1.f + __expf(-(az[g] + (float)gz[g] + bz2)));
      float hold = (float)hc[(size_t)row * H_ + col];
      float n = tanhf((float)gn[g] + bnx + r * (anh[g] + bnh));
      sh_t[wave][g + ((lane >> 4) << 3)][lane & 15] = (1.f - z) * n + z * hold;
    }
    __syncthreads();
    // coalesced writeout: lanes 0-15 -> next-h, lanes 16-31 -> enc (32B each)
    {
      int r = lane & 15;
      v16bf outv;
#pragma unroll
      for (int c2 = 0; c2 < 16; ++c2) outv[c2] = (__bf16)sh_t[wave][r][c2];
      int grow = mt * 16 + r;
      if (lane < 16) {
        *reinterpret_cast<v16bf*>(hn + (size_t)grow * H_ + jt * 16) = outv;
      } else {
        *reinterpret_cast<v16bf*>(enc + ((size_t)grow * T_ + ta) * (2 * H_) + dir * H_ + jt * 16) = outv;
      }
    }
    // cross-WGP step barrier: flush stores, release-add, acquire-spin, invalidate
    __threadfence();
    __syncthreads();
    if (threadIdx.x == 0) {
      __hip_atomic_fetch_add(cdir, 1, __ATOMIC_RELEASE, __HIP_MEMORY_SCOPE_AGENT);
      while (__hip_atomic_load(cdir, __ATOMIC_ACQUIRE, __HIP_MEMORY_SCOPE_AGENT) < (t + 1) * NSPLIT)
        __builtin_amdgcn_s_sleep(1);
    }
    __syncthreads();
    __threadfence();
  }
}

// ---------------- heads: mean and bd (N=24, K=512) ---------------------------
__global__ void heads_kernel(const __bf16* __restrict__ enc,
                             const float* __restrict__ wmean, const float* __restrict__ bmean,
                             const float* __restrict__ wbd, const float* __restrict__ bbd,
                             float* __restrict__ mean_out, float* __restrict__ bdws)
{
  int idx = blockIdx.x * blockDim.x + threadIdx.x;
  const int total = B_ * T_ * 24;
  if (idx >= total) return;
  int o = idx % 24;
  int pos = idx / 24;
  int b = pos / T_, t = pos % T_;
  const __bf16* e = enc + (size_t)pos * (2 * H_);
  const float* w = (o < D_) ? (wmean + (size_t)o * (2 * H_)) : (wbd + (size_t)(o - D_) * (2 * H_));
  float s = (o < D_) ? bmean[o] : bbd[o - D_];
#pragma unroll 8
  for (int k = 0; k < 2 * H_; ++k) s += (float)e[k] * w[k];
  if (o < D_) mean_out[((size_t)b * D_ + o) * T_ + t] = s;   // mean_out [B][D][T]
  else        bdws[(size_t)pos * 16 + (o - D_)] = s;
}

// ---------------- tridiagonal terms + bidiagonal Cholesky --------------------
__global__ void tri_kernel(const float* __restrict__ bdws, float* __restrict__ mainA,
                           float* __restrict__ abA, float* __restrict__ lA,
                           float* __restrict__ subA)
{
  int tid = blockIdx.x * blockDim.x + threadIdx.x;
  if (tid >= B_ * D_) return;
  int b = tid / D_, d = tid % D_;
  float offprev = 0.f, abprev = 0.f, lprev = 1.f;
  for (int t = 0; t < T_; ++t) {
    float x = bdws[((size_t)b * T_ + t) * 16 + d];
    float diag = (x > 0.f) ? (x + log1pf(__expf(-x))) : log1pf(__expf(x));   // softplus
    float off = (t < T_ - 1) ? bdws[((size_t)b * T_ + t) * 16 + D_ + d] : 0.f;
    float mn = diag * diag + offprev * offprev + 1.f + EPS_;
    float ab = diag * off;
    float sub = (t > 0) ? (abprev / lprev) : 0.f;
    float l = sqrtf(mn - sub * sub);
    size_t o = (size_t)tid * T_ + t;
    mainA[o] = mn; abA[o] = ab; lA[o] = l; subA[o] = sub;
    offprev = off; abprev = ab; lprev = l;
  }
}

// ---------------- stream prec + chol with b128 stores ------------------------
__global__ void fill_kernel(const float* __restrict__ mainA, const float* __restrict__ abA,
                            const float* __restrict__ lA, const float* __restrict__ subA,
                            float* __restrict__ out)
{
  const int rows = B_ * D_ * T_;           // 262144 matrix rows
  const int qper = T_ / 4;                 // 64 float4 per row
  const long nmat = (long)rows * qper;     // slots per matrix
  float4* prec4 = reinterpret_cast<float4*>(out + (size_t)B_ * D_ * T_);
  float4* chol4 = prec4 + nmat;
  const long total = 2 * nmat;
  for (long idx = (long)blockIdx.x * blockDim.x + threadIdx.x; idx < total;
       idx += (long)gridDim.x * blockDim.x) {
    long rem = (idx < nmat) ? idx : (idx - nmat);
    int r = (int)(rem / qper);             // (b*D+d)*T + i
    int q = (int)(rem % qper);
    int i = r & (T_ - 1);
    int j0 = q * 4;
    float4 v = make_float4(0.f, 0.f, 0.f, 0.f);
    if (idx < nmat) {                      // prec: tridiagonal
      if (i + 1 >= j0 && i <= j0 + 4) {
        float vals[4];
#pragma unroll
        for (int jj = 0; jj < 4; ++jj) {
          int j = j0 + jj;
          float x = 0.f;
          if (j == i) x = mainA[r];
          else if (j == i - 1) x = abA[r - 1];
          else if (j == i + 1) x = abA[r];
          vals[jj] = x;
        }
        v = make_float4(vals[0], vals[1], vals[2], vals[3]);
      }
      prec4[rem] = v;
    } else {                               // chol: lower bidiagonal
      if (i >= j0 && i <= j0 + 4) {
        float vals[4];
#pragma unroll
        for (int jj = 0; jj < 4; ++jj) {
          int j = j0 + jj;
          float x = 0.f;
          if (j == i) x = lA[r];
          else if (j == i - 1) x = subA[r];
          vals[jj] = x;
        }
        v = make_float4(vals[0], vals[1], vals[2], vals[3]);
      }
      chol4[rem] = v;
    }
  }
}

extern "C" void kernel_launch(void* const* d_in, const int* in_sizes, int n_in,
                              void* d_out, int out_size, void* d_ws, size_t ws_size,
                              hipStream_t stream) {
  const float* y     = (const float*)d_in[0];
  const float* wihf  = (const float*)d_in[1];
  const float* whhf  = (const float*)d_in[2];
  const float* bihf  = (const float*)d_in[3];
  const float* bhhf  = (const float*)d_in[4];
  const float* wihb  = (const float*)d_in[5];
  const float* whhb  = (const float*)d_in[6];
  const float* bihb  = (const float*)d_in[7];
  const float* bhhb  = (const float*)d_in[8];
  const float* wmean = (const float*)d_in[9];
  const float* bmean = (const float*)d_in[10];
  const float* wbd   = (const float*)d_in[11];
  const float* bbd   = (const float*)d_in[12];
  float* out = (float*)d_out;
  char* ws = (char*)d_ws;

  int*    ctr   = (int*)(ws + WS_CTR);
  __bf16* xbf   = (__bf16*)(ws + WS_XBF);
  __bf16* wih   = (__bf16*)(ws + WS_WIH);
  __bf16* whh   = (__bf16*)(ws + WS_WHH);
  __bf16* hbuf  = (__bf16*)(ws + WS_HBUF);
  __bf16* enc   = (__bf16*)(ws + WS_ENC);
  float*  bdws  = (float*)(ws + WS_BD);
  float*  mainA = (float*)(ws + WS_MAIN);
  float*  abA   = (float*)(ws + WS_AB);
  float*  lA    = (float*)(ws + WS_L);
  float*  subA  = (float*)(ws + WS_SUB);
  __bf16* gx    = (__bf16*)(ws + WS_GX);

  const int smem_bytes = 192 * LDW * 2 + 32 * 16 * 17 * 4;   // 101376 + 34816 = 136192

  prep_kernel<<<8192, 256, 0, stream>>>(y, wihf, whhf, wihb, whhb, xbf, wih, whh, hbuf, ctr);
  gx_kernel<<<24576, 256, 0, stream>>>(xbf, wih, gx);        // 196608 tiles, 1 per wave
  gru_recurrence<<<2 * NSPLIT, 1024, smem_bytes, stream>>>(gx, whh, bihf, bhhf, bihb, bhhb,
                                                           hbuf, enc, ctr);
  heads_kernel<<<(B_ * T_ * 24 + 255) / 256, 256, 0, stream>>>(enc, wmean, bmean, wbd, bbd,
                                                               out, bdws);
  tri_kernel<<<4, 256, 0, stream>>>(bdws, mainA, abA, lA, subA);
  fill_kernel<<<16384, 256, 0, stream>>>(mainA, abA, lA, subA, out);
}
